// BottleneckVQ_76424648065079
// MI455X (gfx1250) — matrix operand order repaired and checked by reading, not compiled, not run on hardware
//
#include <hip/hip_runtime.h>

typedef __attribute__((ext_vector_type(16))) _Float16 v16h;
typedef __attribute__((ext_vector_type(8)))  _Float16 v8h;
typedef __attribute__((ext_vector_type(8)))  float    v8f;

#define D_DIM 64
#define K_DIM 512
#define COLS_PER_CHUNK 128
#define COL_PAD 72            // 64 + 8 halves: 144B stride, keeps 16B alignment, spreads banks
#define ROWS_PER_BLOCK 128
#define WAVES_PER_BLOCK 8

// D = A(16x64 rows) x B(64x512 codes); dist = ||e_k||^2 - 2*sim (row norm is constant per row
// and does not affect argmin). f16 hi/lo split recovers ~fp32 accuracy with 3 f16 WMMAs per
// k-chunk instead of 16 fp32 WMMAs.
__global__ __launch_bounds__(256)
void vq_bottleneck_kernel(const float* __restrict__ flat,
                          const float* __restrict__ emb,
                          float* __restrict__ enc,
                          float* __restrict__ outq,
                          int nrows)
{
    __shared__ _Float16 EH[COLS_PER_CHUNK * COL_PAD];  // embeddings hi, [col][d]
    __shared__ _Float16 EL[COLS_PER_CHUNK * COL_PAD];  // embeddings lo (fp32 residual)
    __shared__ float    CN[COLS_PER_CHUNK];            // column norms ||e_k||^2
    __shared__ int      IDX[WAVES_PER_BLOCK * 16];     // argmin per row

    const int  tid    = threadIdx.x;
    const int  lane   = tid & 31;
    const int  wid    = tid >> 5;
    const bool hiHalf = (lane >= 16);

    // ---------------- A fragments: this wave owns rows rbase..rbase+15 ----------------
    const int rbase = blockIdx.x * ROWS_PER_BLOCK + wid * 16;
    int arow = rbase + (lane & 15);
    if (arow >= nrows) arow = nrows - 1;
    const float* ap = flat + (size_t)arow * D_DIM;

    // A 16x32 f16 layout: lane<16 -> K {c0..c0+15} with c0 = kb*32
    //                     lane>=16 -> same + 16
    v16h ah[2], al[2];
    #pragma unroll
    for (int kb = 0; kb < 2; ++kb) {
        const int c0 = kb * 32 + (hiHalf ? 16 : 0);
        float f[16];
        float4* fv = (float4*)f;
        fv[0] = *(const float4*)(ap + c0);
        fv[1] = *(const float4*)(ap + c0 + 4);
        fv[2] = *(const float4*)(ap + c0 + 8);
        fv[3] = *(const float4*)(ap + c0 + 12);
        #pragma unroll
        for (int j = 0; j < 16; ++j) {
            _Float16 h = (_Float16)f[j];
            ah[kb][j] = h;
            al[kb][j] = (_Float16)(f[j] - (float)h);
        }
    }

    float minv[8];
    int   mini[8];
    #pragma unroll
    for (int v = 0; v < 8; ++v) { minv[v] = 3.402823e38f; mini[v] = 0; }

    // ---------------- loop over 4 chunks of 128 codebook columns ----------------
    for (int cc = 0; cc < K_DIM / COLS_PER_CHUNK; ++cc) {
        const int kbase = cc * COLS_PER_CHUNK;

        __syncthreads();   // previous chunk fully consumed before restaging

        // stage 64x128 fp32 -> transposed [col][d] f16 hi/lo in LDS (coalesced float4 reads)
        #pragma unroll
        for (int p = 0; p < 8; ++p) {
            const int vi = p * 256 + tid;   // 2048 float4 per chunk
            const int d  = vi >> 5;         // 0..63
            const int c4 = vi & 31;         // 0..31 (float4 within 128 cols)
            float4 x = *(const float4*)(emb + (size_t)d * K_DIM + kbase + c4 * 4);
            float xs[4] = {x.x, x.y, x.z, x.w};
            const int colb = c4 * 4;
            #pragma unroll
            for (int j = 0; j < 4; ++j) {
                _Float16 h = (_Float16)xs[j];
                EH[(colb + j) * COL_PAD + d] = h;
                EL[(colb + j) * COL_PAD + d] = (_Float16)(xs[j] - (float)h);
            }
        }
        __syncthreads();

        // column norms from staged hi+lo (reconstructs ~22 mantissa bits)
        if (tid < COLS_PER_CHUNK) {
            float s = 0.f;
            const _Float16* cH = EH + tid * COL_PAD;
            const _Float16* cL = EL + tid * COL_PAD;
            #pragma unroll
            for (int d = 0; d < D_DIM; ++d) {
                float e = (float)cH[d] + (float)cL[d];
                s += e * e;
            }
            CN[tid] = s;
        }
        __syncthreads();

        // 8 n-tiles of 16 columns; each wave scans all of them for its 16 rows
        for (int ntl = 0; ntl < 8; ++ntl) {
            const int ncol  = ntl * 16 + (lane & 15);
            const int kglob = kbase + ncol;
            const _Float16* colH = EH + ncol * COL_PAD;
            const _Float16* colL = EL + ncol * COL_PAD;

            v8f acc = {};
            #pragma unroll
            for (int kb = 0; kb < 2; ++kb) {
                // B 32x16 layout: lane<16 holds K kb*32+0..15, lane>=16 holds K kb*32+16..31
                const int kk0 = kb * 32 + (hiHalf ? 16 : 0);
                v8h bh0 = *(const v8h*)(colH + kk0);
                v8h bh1 = *(const v8h*)(colH + kk0 + 8);
                v8h bl0 = *(const v8h*)(colL + kk0);
                v8h bl1 = *(const v8h*)(colL + kk0 + 8);
                v16h bh = __builtin_shufflevector(bh0, bh1,
                            0,1,2,3,4,5,6,7,8,9,10,11,12,13,14,15);
                v16h bl = __builtin_shufflevector(bl0, bl1,
                            0,1,2,3,4,5,6,7,8,9,10,11,12,13,14,15);
                acc = __builtin_amdgcn_wmma_f32_16x16x32_f16(
                          false, ah[kb], false, bh, (short)0, acc, false, false);
                acc = __builtin_amdgcn_wmma_f32_16x16x32_f16(
                          false, ah[kb], false, bl, (short)0, acc, false, false);
                acc = __builtin_amdgcn_wmma_f32_16x16x32_f16(
                          false, al[kb], false, bh, (short)0, acc, false, false);
            }

            const float cn = CN[ncol];
            #pragma unroll
            for (int v = 0; v < 8; ++v) {
                float dval = cn - 2.0f * acc[v];       // row norm omitted (constant per row)
                if (dval < minv[v]) { minv[v] = dval; mini[v] = kglob; }
            }
        }
    }

    // ---------------- cross-lane argmin within each 16-lane half ----------------
    // C layout: lanes 0..15 hold rows 0..7 (vgpr v), lanes 16..31 hold rows 8..15.
    #pragma unroll
    for (int off = 1; off <= 8; off <<= 1) {
        #pragma unroll
        for (int v = 0; v < 8; ++v) {
            float ov = __shfl_xor(minv[v], off, 32);
            int   oi = __shfl_xor(mini[v], off, 32);
            if (ov < minv[v] || (ov == minv[v] && oi < mini[v])) {
                minv[v] = ov; mini[v] = oi;            // tie -> lowest index (argmin semantics)
            }
        }
    }
    if (lane == 0) {
        #pragma unroll
        for (int v = 0; v < 8; ++v) IDX[wid * 16 + v] = mini[v];
    }
    if (lane == 16) {
        #pragma unroll
        for (int v = 0; v < 8; ++v) IDX[wid * 16 + 8 + v] = mini[v];
    }
    __syncthreads();

    // ---------------- outputs: one-hot encodings + gathered fp32 code vectors ----------------
    #pragma unroll 1
    for (int m = 0; m < 16; ++m) {
        const int r = rbase + m;
        if (r >= nrows) break;
        const int k = IDX[wid * 16 + m];

        // quantized = emb[:, k] (fp32 from global; 128 KB table is L2-resident)
        outq[(size_t)r * D_DIM + lane]      = emb[(size_t)lane * K_DIM + k];
        outq[(size_t)r * D_DIM + 32 + lane] = emb[(size_t)(lane + 32) * K_DIM + k];

        // one-hot row, coalesced b128 stores (dominant HBM traffic: 128 MB total)
        float* er = enc + (size_t)r * K_DIM;
        #pragma unroll
        for (int it = 0; it < 4; ++it) {
            const int c = it * 128 + lane * 4;
            float4 vz;
            vz.x = (float)(c     == k);
            vz.y = (float)(c + 1 == k);
            vz.z = (float)(c + 2 == k);
            vz.w = (float)(c + 3 == k);
            *(float4*)(er + c) = vz;
        }
    }
}

extern "C" void kernel_launch(void* const* d_in, const int* in_sizes, int n_in,
                              void* d_out, int out_size, void* d_ws, size_t ws_size,
                              hipStream_t stream) {
    (void)n_in; (void)out_size; (void)d_ws; (void)ws_size;
    const float* flat = (const float*)d_in[0];   // batch  [B,H,W,D] fp32
    const float* emb  = (const float*)d_in[1];   // embeddings [D,K] fp32

    const int nrows = in_sizes[0] / D_DIM;       // 65536
    float* out  = (float*)d_out;
    float* enc  = out;                           // [N, K]
    float* outq = out + (size_t)nrows * K_DIM;   // [N, D] (== [B,H,W,D])

    const int grid = (nrows + ROWS_PER_BLOCK - 1) / ROWS_PER_BLOCK;  // 512
    vq_bottleneck_kernel<<<grid, 256, 0, stream>>>(flat, emb, enc, outq, nrows);
}
